// SSMDecoder_55997783605678
// MI455X (gfx1250) — matrix operand order repaired and checked by reading, not compiled, not run on hardware
//
#include <hip/hip_runtime.h>
#include <hip/hip_bf16.h>
#include <cstdint>

// Problem dims (match reference)
#define B_   4
#define T_   4096
#define D_   1024
#define S_   64
#define L_   4
#define V_   8192
#define CD_  512
#define NC_  16
#define M_   (B_ * T_)      // 16384 rows
#define NCH  16             // scan chunks
#define CHUNK 256           // T_ / NCH
#define EPSLN 1e-5f

typedef unsigned short u16;
typedef __bf16 bf16;
typedef __attribute__((ext_vector_type(16))) __bf16 v16bf;
typedef __attribute__((ext_vector_type(8)))  float  v8f;
typedef unsigned int v4u __attribute__((ext_vector_type(4)));
typedef int          v8i __attribute__((ext_vector_type(8)));
typedef int          v4i __attribute__((ext_vector_type(4)));

#if defined(__has_builtin)
#  if __has_builtin(__builtin_amdgcn_tensor_load_to_lds)
#    define HAVE_TDM 1
#  else
#    define HAVE_TDM 0
#  endif
#else
#  define HAVE_TDM 0
#endif

union FragAB { uint4 q[2]; v16bf v; };

__device__ __forceinline__ u16 f2bf(float f) {
  union { bf16 h; u16 u; } c; c.h = (bf16)f; return c.u;
}

// Low 32 bits of a generic pointer into __shared__ = LDS byte offset.
__device__ __forceinline__ uint32_t lds_addr_of(const void* p) {
  return (uint32_t)(uintptr_t)p;
}

// Per-lane async DMA: LDS[lds] = MEM[gaddr], 16 bytes. Tracked by ASYNCcnt.
__device__ __forceinline__ void async_load_b128(uint32_t lds, unsigned long long gaddr) {
  asm volatile("global_load_async_to_lds_b128 %0, %1, off"
               :: "v"(lds), "v"(gaddr) : "memory");
}
__device__ __forceinline__ void wait_asynccnt0() {
  asm volatile("s_wait_asynccnt 0x0" ::: "memory");
}

// A fragment: 16x32 bf16, source row-major [M x ld] (K contiguous).
// ISA 7.12.2: lanes 0-15 -> M=lane, K base 0/16; lanes 16-31 -> M=lane-16, K base 8/24.
__device__ __forceinline__ void load_frag_a(const u16* base, int ld, int m0, int k0,
                                            int lane, FragAB& f) {
  const int m  = m0 + (lane & 15);
  const int hi = lane >> 4;
  const u16* p = base + (size_t)m * ld + k0 + hi * 8;
  f.q[0] = *reinterpret_cast<const uint4*>(p);
  f.q[1] = *reinterpret_cast<const uint4*>(p + 16);
}

// B fragment: 32x16 bf16 (KxN), source stored row-major [N x ld] (B^T, K contiguous).
// Lanes 0-15 hold K=0..7,16..23 of column N=lane; lanes 16-31 hold K=8..15,24..31.
__device__ __forceinline__ void load_frag_b(const u16* base, int ld, int n0, int k0,
                                            int lane, FragAB& f) {
  const int n  = n0 + (lane & 15);
  const int hi = lane >> 4;
  const u16* p = base + (size_t)n * ld + k0 + hi * 8;
  f.q[0] = *reinterpret_cast<const uint4*>(p);
  f.q[1] = *reinterpret_cast<const uint4*>(p + 16);
}

__device__ __forceinline__ v8f wmma_bf16(const FragAB& a, const FragAB& b, v8f c) {
  return __builtin_amdgcn_wmma_f32_16x16x32_bf16(false, a.v, false, b.v,
                                                 (short)0, c, false, false);
}

// ---------------------------------------------------------------------------
// K0: f32 -> bf16 conversion (weights)
__global__ void k_f32_to_bf16(const float* __restrict__ in, u16* __restrict__ out, int n) {
  int i = blockIdx.x * blockDim.x + threadIdx.x;
  const int stride = gridDim.x * blockDim.x;
  for (; i < n; i += stride) out[i] = f2bf(in[i]);
}

// ---------------------------------------------------------------------------
// K1: x = embed[tok] + pos_embed[t]  (f32 + bf16 copies)
__global__ void __launch_bounds__(256) k_embed(const int* __restrict__ tokens,
                                               const float* __restrict__ embed,
                                               const float* __restrict__ pos,
                                               float* __restrict__ xf,
                                               u16* __restrict__ xh) {
  const int m = blockIdx.x;
  const int t = m & (T_ - 1);
  const int tok = tokens[m];
  const float4* e = reinterpret_cast<const float4*>(embed + (size_t)tok * D_);
  const float4* p = reinterpret_cast<const float4*>(pos + (size_t)t * D_);
  float4* xo = reinterpret_cast<float4*>(xf + (size_t)m * D_);
  u16* ho = xh + (size_t)m * D_;
  for (int i = threadIdx.x; i < D_ / 4; i += blockDim.x) {
    const float4 a = e[i], b = p[i];
    const float4 r = make_float4(a.x + b.x, a.y + b.y, a.z + b.z, a.w + b.w);
    xo[i] = r;
    ushort4 hv;
    hv.x = f2bf(r.x); hv.y = f2bf(r.y); hv.z = f2bf(r.z); hv.w = f2bf(r.w);
    *reinterpret_cast<ushort4*>(ho + i * 4) = hv;
  }
}

// ---------------------------------------------------------------------------
// K2: h0[s] = c2h_b[s] + sum_c c2h_w[s,c] * mean_n(c_states[n,c])
__global__ void k_h0(const float* __restrict__ cs, const float* __restrict__ w,
                     const float* __restrict__ b, float* __restrict__ h0) {
  __shared__ float pooled[CD_];
  for (int c = threadIdx.x; c < CD_; c += blockDim.x) {
    float s = 0.f;
    for (int n = 0; n < NC_; ++n) s += cs[n * CD_ + c];
    pooled[c] = s * (1.0f / NC_);
  }
  __syncthreads();
  if (threadIdx.x < S_) {
    const int s = threadIdx.x;
    float acc = b[s];
    const float* wr = w + (size_t)s * CD_;
    for (int c = 0; c < CD_; ++c) acc += wr[c] * pooled[c];
    h0[s] = acc;
  }
}

// ---------------------------------------------------------------------------
// K3: bts = (x @ B_w^T + B_b) * dt   — WMMA GEMM [M,1024]x[1024,64]
// Whole B_w[l] slab (64x1024 bf16 = 128 KB) is DMA'd into LDS once per WG
// (TDM tensor_load_to_lds when available, else async-to-LDS). The K-loop
// double-buffers complete fragment sets in registers so the 8 ds_loads of
// step k+1 overlap the 4 WMMAs of step k (no per-WMMA dscnt waits / WAR NOPs).
__global__ void __launch_bounds__(256) k_bproj(const u16* __restrict__ x,
                                               const u16* __restrict__ bw,
                                               const float* __restrict__ bb,
                                               const float* __restrict__ logdt,
                                               float* __restrict__ bts) {
  extern __shared__ u16 ldsB[];   // 64 * 1024 bf16 = 131072 bytes
  const int tid = threadIdx.x, lane = tid & 31, wave = tid >> 5;

#if HAVE_TDM
  if (wave == 0) {
    const uint32_t lds = lds_addr_of(ldsB);
    const unsigned long long ga = (unsigned long long)(uintptr_t)bw;
    v4u g0;
    g0[0] = 1u;                                            // count=1 (valid D#)
    g0[1] = lds;                                           // lds_addr
    g0[2] = (uint32_t)ga;                                  // global_addr[31:0]
    g0[3] = (uint32_t)((ga >> 32) & 0x1FFFFFFull) | (2u << 30); // addr[56:32] | type=2
    v8i g1;
    g1[0] = (int)(1u << 16);                               // data_size=1 (2 bytes)
    g1[1] = (int)((D_ & 0xFFFFu) << 16);                   // tensor_dim0[15:0]
    g1[2] = (int)((D_ >> 16) | ((S_ & 0xFFFFu) << 16));    // dim0 hi | tensor_dim1 lo
    g1[3] = (int)((S_ >> 16) | ((unsigned)D_ << 16));      // dim1 hi | tile_dim0=1024
    g1[4] = (int)S_;                                       // tile_dim1=64, tile_dim2=0
    g1[5] = (int)D_;                                       // tensor_dim0_stride lo32
    g1[6] = 0;                                             // stride hi | dim1_stride lo
    g1[7] = 0;
    v4i gz = {0, 0, 0, 0};
#if __clang_major__ >= 23
    v8i gz8 = {0, 0, 0, 0, 0, 0, 0, 0};
    __builtin_amdgcn_tensor_load_to_lds(g0, g1, gz, gz, gz8, 0);
#else
    __builtin_amdgcn_tensor_load_to_lds(g0, g1, gz, gz, 0);
#endif
    __builtin_amdgcn_s_wait_tensorcnt(0);
  }
#else
  {
    const uint32_t ldsbase = lds_addr_of(ldsB);
    const unsigned long long g = (unsigned long long)(uintptr_t)bw;
    for (int byte = tid * 16; byte < (int)(S_ * D_ * sizeof(u16)); byte += 256 * 16)
      async_load_b128(ldsbase + (uint32_t)byte, g + (unsigned long long)byte);
    wait_asynccnt0();
  }
#endif
  __syncthreads();

  const int m0 = blockIdx.x * 128 + wave * 16;
  v8f acc[4] = {};
  FragAB fa[2], fb[2][4];

  load_frag_a(x, D_, m0, 0, lane, fa[0]);
#pragma unroll
  for (int nt = 0; nt < 4; ++nt) load_frag_b(ldsB, D_, nt * 16, 0, lane, fb[0][nt]);

  for (int k = 0; k < D_; k += 32) {
    const int cur = (k >> 5) & 1, nxt = cur ^ 1;
    if (k + 32 < D_) {
      load_frag_a(x, D_, m0, k + 32, lane, fa[nxt]);
#pragma unroll
      for (int nt = 0; nt < 4; ++nt) load_frag_b(ldsB, D_, nt * 16, k + 32, lane, fb[nxt][nt]);
    }
#pragma unroll
    for (int nt = 0; nt < 4; ++nt)
      acc[nt] = wmma_bf16(fa[cur], fb[cur][nt], acc[nt]);
  }

  const int hi = lane >> 4;
#pragma unroll
  for (int nt = 0; nt < 4; ++nt) {
    const int s = nt * 16 + (lane & 15);
    const float dts = fminf(__expf(logdt[s]), 1.0f);
    const float bias = bb[s];
#pragma unroll
    for (int j = 0; j < 8; ++j) {
      const int m = m0 + j + hi * 8;
      bts[(size_t)m * S_ + s] = (acc[nt][j] + bias) * dts;
    }
  }
}

// ---------------------------------------------------------------------------
// K4: chunked scan, pass A: per-(b,s,chunk) reduce with h=0 start
__global__ void k_scan_reduce(const float* __restrict__ bts,
                              const float* __restrict__ logA,
                              const float* __restrict__ logdt,
                              float* __restrict__ carry) {
  const int gid = blockIdx.x * blockDim.x + threadIdx.x;   // 0..4095
  const int s  = gid & (S_ - 1);
  const int ch = (gid >> 6) & (NCH - 1);
  const int b  = gid >> 10;
  const float dt = fminf(__expf(logdt[s]), 1.0f);
  const float A  = __expf(-__expf(logA[s]) * dt);
  const float* p = bts + ((size_t)(b * T_ + ch * CHUNK)) * S_ + s;
  float h = 0.f;
  for (int i = 0; i < CHUNK; ++i) {
    if ((i & 15) == 0) __builtin_prefetch(p + (size_t)(i + 32) * S_, 0, 1);
    h = fmaf(A, h, p[(size_t)i * S_]);
  }
  carry[(b * NCH + ch) * S_ + s] = h;
}

// K5: pass B: prefix over the 16 chunks (A^256 known analytically)
__global__ void k_scan_prefix(const float* __restrict__ carry,
                              const float* __restrict__ h0,
                              const float* __restrict__ logA,
                              const float* __restrict__ logdt,
                              float* __restrict__ start) {
  const int tid = threadIdx.x;             // 256 threads = B*S channels
  const int s = tid & (S_ - 1), b = tid >> 6;
  const float dt   = fminf(__expf(logdt[s]), 1.0f);
  const float a256 = __expf(-__expf(logA[s]) * dt * (float)CHUNK);
  float h = h0[s];
  for (int j = 0; j < NCH; ++j) {
    start[(b * NCH + j) * S_ + s] = h;
    h = fmaf(a256, h, carry[(b * NCH + j) * S_ + s]);
  }
}

// K6: pass C: replay each chunk from its true start and emit hs (bf16)
__global__ void k_scan_emit(const float* __restrict__ bts,
                            const float* __restrict__ start,
                            const float* __restrict__ logA,
                            const float* __restrict__ logdt,
                            u16* __restrict__ hs) {
  const int gid = blockIdx.x * blockDim.x + threadIdx.x;
  const int s  = gid & (S_ - 1);
  const int ch = (gid >> 6) & (NCH - 1);
  const int b  = gid >> 10;
  const float dt = fminf(__expf(logdt[s]), 1.0f);
  const float A  = __expf(-__expf(logA[s]) * dt);
  const size_t base = ((size_t)(b * T_ + ch * CHUNK)) * S_ + s;
  const float* p = bts + base;
  u16* o = hs + base;
  float h = start[(b * NCH + ch) * S_ + s];
  for (int i = 0; i < CHUNK; ++i) {
    if ((i & 15) == 0) __builtin_prefetch(p + (size_t)(i + 32) * S_, 0, 1);
    h = fmaf(A, h, p[(size_t)i * S_]);
    o[(size_t)i * S_] = f2bf(h);
  }
}

// ---------------------------------------------------------------------------
// K7: y = hs @ C_w^T + C_b + D*u ; x = LayerNorm(x + y)   (fully fused)
// WG covers 16 rows x all 1024 cols: 8 waves x 8 N-tiles, K=64 (2 steps).
// All 9 fragment loads of a K-step are batched ahead of the 8-WMMA chain.
__global__ void __launch_bounds__(256) k_cproj_ln(const u16* __restrict__ hs,
                                                  const u16* __restrict__ cw,
                                                  const float* __restrict__ cb,
                                                  const float* __restrict__ dp,
                                                  const float* __restrict__ lw,
                                                  const float* __restrict__ lb,
                                                  float* __restrict__ xf,
                                                  u16* __restrict__ xh) {
  __shared__ float redsum[16], redsq[16], mu[16], rstd[16];
  const int tid = threadIdx.x, lane = tid & 31, wave = tid >> 5, hi = lane >> 4;
  const int m0 = blockIdx.x * 16;
  if (tid < 16) { redsum[tid] = 0.f; redsq[tid] = 0.f; }

  v8f acc[8] = {};
#pragma unroll
  for (int ks = 0; ks < 2; ++ks) {
    const int k = ks * 32;
    FragAB fa, fb[8];
    load_frag_a(hs, S_, m0, k, lane, fa);
#pragma unroll
    for (int jt = 0; jt < 8; ++jt)
      load_frag_b(cw, S_, (wave * 8 + jt) * 16, k, lane, fb[jt]);
#pragma unroll
    for (int jt = 0; jt < 8; ++jt)
      acc[jt] = wmma_bf16(fa, fb[jt], acc[jt]);
  }

  // epilogue: pre = acc + C_b + (1 + D) * u  (u = residual input)
  float lwv[8], lbv[8];
  float psum[8] = {0,0,0,0,0,0,0,0}, psq[8] = {0,0,0,0,0,0,0,0};
#pragma unroll
  for (int jt = 0; jt < 8; ++jt) {
    const int n = (wave * 8 + jt) * 16 + (lane & 15);
    const float cbn = cb[n], dpn = dp[n];
    lwv[jt] = lw[n]; lbv[jt] = lb[n];
#pragma unroll
    for (int j = 0; j < 8; ++j) {
      const int m = m0 + j + hi * 8;
      const float u = xf[(size_t)m * D_ + n];
      const float pre = acc[jt][j] + cbn + u + dpn * u;
      acc[jt][j] = pre;
      psum[j] += pre;
      psq[j]  += pre * pre;
    }
  }
  __syncthreads();   // redsum/redsq init visible
#pragma unroll
  for (int j = 0; j < 8; ++j) {
    const int row = j + hi * 8;
    atomicAdd(&redsum[row], psum[j]);
    atomicAdd(&redsq[row],  psq[j]);
  }
  __syncthreads();
  if (tid < 16) {
    const float m  = redsum[tid] * (1.0f / D_);
    const float vv = redsq[tid]  * (1.0f / D_) - m * m;
    mu[tid]   = m;
    rstd[tid] = rsqrtf(vv + EPSLN);
  }
  __syncthreads();
#pragma unroll
  for (int jt = 0; jt < 8; ++jt) {
    const int n = (wave * 8 + jt) * 16 + (lane & 15);
#pragma unroll
    for (int j = 0; j < 8; ++j) {
      const int row = j + hi * 8;
      const int m = m0 + row;
      const float v = (acc[jt][j] - mu[row]) * rstd[row] * lwv[jt] + lbv[jt];
      xf[(size_t)m * D_ + n] = v;
      xh[(size_t)m * D_ + n] = f2bf(v);
    }
  }
}

// ---------------------------------------------------------------------------
// K8: final LayerNorm -> bf16 activations for the head GEMM
__global__ void __launch_bounds__(256) k_lnf(const float* __restrict__ xf,
                                             const float* __restrict__ w,
                                             const float* __restrict__ b,
                                             u16* __restrict__ out) {
  __shared__ float ps[8], pq[8], stat[2];
  const int m = blockIdx.x, tid = threadIdx.x, lane = tid & 31, wave = tid >> 5;
  const float* row = xf + (size_t)m * D_;
  float s = 0.f, q = 0.f, v[4];
#pragma unroll
  for (int i = 0; i < 4; ++i) {
    v[i] = row[tid + i * 256];
    s += v[i]; q += v[i] * v[i];
  }
#pragma unroll
  for (int o = 16; o > 0; o >>= 1) {
    s += __shfl_xor(s, o, 32);
    q += __shfl_xor(q, o, 32);
  }
  if (lane == 0) { ps[wave] = s; pq[wave] = q; }
  __syncthreads();
  if (tid == 0) {
    float S = 0.f, Q = 0.f;
    for (int i = 0; i < 8; ++i) { S += ps[i]; Q += pq[i]; }
    const float mm = S * (1.0f / D_);
    stat[0] = mm;
    stat[1] = rsqrtf(Q * (1.0f / D_) - mm * mm + EPSLN);
  }
  __syncthreads();
  const float mm = stat[0], rs = stat[1];
  u16* orow = out + (size_t)m * D_;
#pragma unroll
  for (int i = 0; i < 4; ++i) {
    const int d = tid + i * 256;
    orow[d] = f2bf((v[i] - mm) * rs * w[d] + b[d]);
  }
}

// ---------------------------------------------------------------------------
// K9: logits = x_ln @ head_w^T   [16384,1024]x[1024,8192], f32 out.
// 128x128 WG tile, 8 waves (2x4), 64x32 per wave. A/B double buffers staged in
// LDS with global_load_async_to_lds_b128 (ASYNCcnt), non-temporal f32 stores.
__global__ void __launch_bounds__(256) k_head(const u16* __restrict__ x,
                                              const u16* __restrict__ hw,
                                              float* __restrict__ out) {
  __shared__ alignas(16) u16 sA[2][128 * 32];
  __shared__ alignas(16) u16 sB[2][128 * 32];
  const int tid = threadIdx.x, lane = tid & 31, wave = tid >> 5, hi = lane >> 4;
  const int mb = blockIdx.x & 127, nb = blockIdx.x >> 7;
  const int m_base = mb * 128, n_base = nb * 128;
  const int wm = wave & 1, wn = wave >> 1;
  const int ldrow = tid >> 1, ldcol = (tid & 1) * 16;
  const unsigned long long gAa =
      (unsigned long long)(uintptr_t)(x + (size_t)(m_base + ldrow) * D_ + ldcol);
  const unsigned long long gBa =
      (unsigned long long)(uintptr_t)(hw + (size_t)(n_base + ldrow) * D_ + ldcol);
  const int soff = ldrow * 32 + ldcol;
  const uint32_t sAoff[2] = { lds_addr_of(&sA[0][soff]), lds_addr_of(&sA[1][soff]) };
  const uint32_t sBoff[2] = { lds_addr_of(&sB[0][soff]), lds_addr_of(&sB[1][soff]) };

  auto issue_stage = [&](int ks, int buf) {
    const unsigned long long kb = (unsigned long long)ks * 64ull;  // 32 elems * 2B
    async_load_b128(sAoff[buf],      gAa + kb);
    async_load_b128(sAoff[buf] + 16, gAa + kb + 16);
    async_load_b128(sBoff[buf],      gBa + kb);
    async_load_b128(sBoff[buf] + 16, gBa + kb + 16);
  };

  v8f acc[4][2] = {};
  issue_stage(0, 0);
  wait_asynccnt0();
  __syncthreads();

  for (int ks = 0; ks < 32; ++ks) {
    const int cur = ks & 1;
    if (ks + 1 < 32) issue_stage(ks + 1, cur ^ 1);   // overlap DMA with math
    FragAB fa[4], fb[2];
#pragma unroll
    for (int i = 0; i < 4; ++i) load_frag_a(sA[cur], 32, wm * 64 + i * 16, 0, lane, fa[i]);
#pragma unroll
    for (int j = 0; j < 2; ++j) load_frag_b(sB[cur], 32, wn * 32 + j * 16, 0, lane, fb[j]);
#pragma unroll
    for (int i = 0; i < 4; ++i)
#pragma unroll
      for (int j = 0; j < 2; ++j)
        acc[i][j] = wmma_bf16(fa[i], fb[j], acc[i][j]);
    if (ks + 1 < 32) wait_asynccnt0();
    __syncthreads();
  }

#pragma unroll
  for (int i = 0; i < 4; ++i) {
#pragma unroll
    for (int j = 0; j < 2; ++j) {
      const int n = n_base + wn * 32 + j * 16 + (lane & 15);
      const int mrow = m_base + wm * 64 + i * 16 + hi * 8;
#pragma unroll
      for (int jj = 0; jj < 8; ++jj)
        __builtin_nontemporal_store(acc[i][j][jj], &out[(size_t)(mrow + jj) * V_ + n]);
    }
  }
}

// ---------------------------------------------------------------------------
extern "C" void kernel_launch(void* const* d_in, const int* in_sizes, int n_in,
                              void* d_out, int out_size, void* d_ws, size_t ws_size,
                              hipStream_t stream) {
  (void)in_sizes; (void)n_in; (void)out_size; (void)ws_size;
  const int*   tokens   = (const int*)d_in[0];
  const float* c_states = (const float*)d_in[1];
  const float* embed    = (const float*)d_in[2];
  const float* pos      = (const float*)d_in[3];
  const float* logA     = (const float*)d_in[4];
  const float* logdt    = (const float*)d_in[5];
  const float* B_w      = (const float*)d_in[6];
  const float* B_b      = (const float*)d_in[7];
  const float* C_w      = (const float*)d_in[8];
  const float* C_b      = (const float*)d_in[9];
  const float* D_p      = (const float*)d_in[10];
  const float* ln_w     = (const float*)d_in[11];
  const float* ln_b     = (const float*)d_in[12];
  const float* c2h_w    = (const float*)d_in[13];
  const float* c2h_b    = (const float*)d_in[14];
  const float* lnf_w    = (const float*)d_in[15];
  const float* lnf_b    = (const float*)d_in[16];
  const float* head_w   = (const float*)d_in[17];
  float* out = (float*)d_out;

  char* ws = (char*)d_ws;
  size_t off = 0;
  auto alloc = [&](size_t bytes) -> void* {
    void* p = ws + off;
    off = (off + bytes + 255) & ~(size_t)255;
    return p;
  };
  float* xf    = (float*)alloc((size_t)M_ * D_ * 4);   // activations f32
  u16*   xh    = (u16*)  alloc((size_t)M_ * D_ * 2);   // activations bf16
  float* bts   = (float*)alloc((size_t)M_ * S_ * 4);   // discretized inputs
  u16*   hs    = (u16*)  alloc((size_t)M_ * S_ * 2);   // scan states bf16
  float* h0    = (float*)alloc(S_ * 4);
  float* carry = (float*)alloc((size_t)B_ * NCH * S_ * 4);
  float* start = (float*)alloc((size_t)B_ * NCH * S_ * 4);
  u16*   bw_h  = (u16*)  alloc((size_t)L_ * S_ * D_ * 2);
  u16*   cw_h  = (u16*)  alloc((size_t)L_ * D_ * S_ * 2);
  u16*   hw_h  = (u16*)  alloc((size_t)V_ * D_ * 2);

  k_f32_to_bf16<<<2048, 256, 0, stream>>>(head_w, hw_h, V_ * D_);
  k_f32_to_bf16<<<512,  256, 0, stream>>>(B_w,    bw_h, L_ * S_ * D_);
  k_f32_to_bf16<<<512,  256, 0, stream>>>(C_w,    cw_h, L_ * D_ * S_);
  k_embed<<<M_, 256, 0, stream>>>(tokens, embed, pos, xf, xh);
  k_h0<<<1, 256, 0, stream>>>(c_states, c2h_w, c2h_b, h0);

  for (int l = 0; l < L_; ++l) {
    const u16* bw = bw_h + (size_t)l * S_ * D_;
    const u16* cw = cw_h + (size_t)l * D_ * S_;
    const float* la = logA + l * S_;
    const float* ld = logdt + l * S_;
    k_bproj<<<M_ / 128, 256, S_ * D_ * sizeof(u16), stream>>>(xh, bw, B_b + l * S_, ld, bts);
    k_scan_reduce<<<16, 256, 0, stream>>>(bts, la, ld, carry);
    k_scan_prefix<<<1, 256, 0, stream>>>(carry, h0, la, ld, start);
    k_scan_emit<<<16, 256, 0, stream>>>(bts, start, la, ld, hs);
    k_cproj_ln<<<M_ / 16, 256, 0, stream>>>(hs, cw, C_b + l * D_, D_p + l * D_,
                                            ln_w + l * D_, ln_b + l * D_, xf, xh);
  }

  k_lnf<<<M_, 256, 0, stream>>>(xf, lnf_w, lnf_b, xh);
  k_head<<<(M_ / 128) * (V_ / 128), 256, 0, stream>>>(xh, hw_h, out);
}
